// MultiheadAttention_3693671874676
// MI455X (gfx1250) — matrix operand order repaired and checked by reading, not compiled, not run on hardware
//
#include <hip/hip_runtime.h>
#include <math.h>

// ---------------------------------------------------------------------------
// MultiheadAttention forward for MI455X (gfx1250, wave32, WMMA 16x16x32 bf16)
// B=20, S=1024, D_MODEL=768, H=6, DK=128.
// d_out = [ out(20,1024,768) f32 | attn(20,6,1024,1024) f32 ]
// Attention is a fused two-pass flash kernel: raw scores never hit HBM.
// ---------------------------------------------------------------------------

typedef __attribute__((ext_vector_type(16))) __bf16 v16bf;
typedef __attribute__((ext_vector_type(8)))  float  v8f;

// Native f32 -> bf16 (RNE) through the __bf16 type: lets the backend use the
// hardware convert (and fuse pairs into v_cvt_pk_bf16_f32) instead of the
// and/add3/shift bit-twiddle chain.
__device__ __forceinline__ unsigned short f2bf(float f) {
  union { __bf16 b; unsigned short u; } c; c.b = (__bf16)f; return c.u;
}
__device__ __forceinline__ unsigned pk_bf16(float lo, float hi) {
  union { __bf16 b[2]; unsigned u; } c;
  c.b[0] = (__bf16)lo; c.b[1] = (__bf16)hi; return c.u;
}

union BFrag { v16bf v; unsigned int u32[8]; };

// A-matrix 16x32 bf16 fragment from an LDS tile with row stride `stride`
// (elements), K-chunk offset `koff`. ISA layout: lanes 0-15 row=lane,
// VGPR0-3 K=0..7, VGPR4-7 K=16..23; lanes 16-31 same rows, K offset +8.
__device__ __forceinline__ v16bf load_afrag_s(const unsigned short* As, int mbase,
                                              int lane, int stride, int koff) {
  BFrag f;
  const int row = mbase + (lane & 15);
  const int khi = (lane >> 4) << 3;
  const unsigned int* rowp = (const unsigned int*)(As + row * stride + koff);
#pragma unroll
  for (int j = 0; j < 8; ++j) {
    const int kk = khi + ((j & 4) << 2) + ((j & 3) << 1);
    f.u32[j] = rowp[kk >> 1];
  }
  return f.v;
}

// B-matrix 32x16 bf16 fragment from an LDS tile stored [N][K] with row stride
// `stride`. ISA layout: N = lane&15; VGPR0-3 K=0..7, VGPR4-7 K=16..23; lanes 16-31 K offset +8.
__device__ __forceinline__ v16bf load_bfrag_s(const unsigned short* Bs, int nbase,
                                              int lane, int stride, int koff) {
  BFrag f;
  const int col = nbase + (lane & 15);
  const int khi = (lane >> 4) << 3;
  const unsigned int* colp = (const unsigned int*)(Bs + col * stride + koff);
#pragma unroll
  for (int j = 0; j < 8; ++j) {
    const int kk = khi + ((j & 4) << 2) + ((j & 3) << 1);
    f.u32[j] = colp[kk >> 1];
  }
  return f.v;
}

__device__ __forceinline__ v8f wmma_bf16(v16bf a, v16bf b, v8f c) {
  return __builtin_amdgcn_wmma_f32_16x16x32_bf16(false, a, false, b, (short)0, c, false, false);
}

// ---------------------------------------------------------------------------
// Kernel 1: projection GEMM  P = X[20480x768] @ W[768x768] -> bf16, head-major
// [B][H][S][128]; `scale` folds in Q's 1/sqrt(Dk). grid (160,6), block 256.
// ---------------------------------------------------------------------------
__global__ __launch_bounds__(256) void proj_gemm(const float* __restrict__ X,
                                                 const float* __restrict__ W,
                                                 unsigned short* __restrict__ P,
                                                 float scale) {
  __shared__ __attribute__((aligned(16))) unsigned short As[128 * 32];
  __shared__ __attribute__((aligned(16))) unsigned short Bs[128 * 32];
  const int tid = threadIdx.x, lane = tid & 31, wave = tid >> 5;
  const int m0 = blockIdx.x * 128, n0 = blockIdx.y * 128;
  v8f acc[8] = {};
  for (int kt = 0; kt < 768; kt += 32) {
#pragma unroll
    for (int i = tid; i < 1024; i += 256) {          // A: 128x32, float4 loads
      const int row = i >> 3, c4 = (i & 7) << 2;
      const float4 xv = *(const float4*)(X + (size_t)(m0 + row) * 768 + kt + c4);
      uint2 pk;
      pk.x = pk_bf16(xv.x, xv.y);
      pk.y = pk_bf16(xv.z, xv.w);
      *(uint2*)(As + row * 32 + c4) = pk;
    }
#pragma unroll
    for (int i = tid; i < 512; i += 256) {           // B -> [N][K], k-pair packed
      const int kk = (i >> 5) << 1, n4 = (i & 31) << 2;
      const float4 w0 = *(const float4*)(W + (size_t)(kt + kk) * 768 + n0 + n4);
      const float4 w1 = *(const float4*)(W + (size_t)(kt + kk + 1) * 768 + n0 + n4);
      *(unsigned*)(Bs + (n4 + 0) * 32 + kk) = pk_bf16(w0.x, w1.x);
      *(unsigned*)(Bs + (n4 + 1) * 32 + kk) = pk_bf16(w0.y, w1.y);
      *(unsigned*)(Bs + (n4 + 2) * 32 + kk) = pk_bf16(w0.z, w1.z);
      *(unsigned*)(Bs + (n4 + 3) * 32 + kk) = pk_bf16(w0.w, w1.w);
    }
    if (kt + 32 < 768)                               // global_prefetch_b8
      __builtin_prefetch(X + (size_t)(m0 + (tid >> 1)) * 768 + kt + 32 + ((tid & 1) << 4), 0, 0);
    __syncthreads();
    const v16bf a = load_afrag_s(As, wave * 16, lane, 32, 0);
#pragma unroll
    for (int nf = 0; nf < 8; ++nf) {
      const v16bf b = load_bfrag_s(Bs, nf * 16, lane, 32, 0);
      acc[nf] = wmma_bf16(a, b, acc[nf]);
    }
    __syncthreads();
  }
#pragma unroll
  for (int nf = 0; nf < 8; ++nf) {
    union { v8f v; float f[8]; } t; t.v = acc[nf];
#pragma unroll
    for (int r = 0; r < 8; ++r) {                    // C: row=r(+8 hi), N=lane&15
      const int m = m0 + wave * 16 + r + ((lane >> 4) << 3);
      const int n = n0 + nf * 16 + (lane & 15);
      const int b = m >> 10, s = m & 1023, h = n >> 7, d = n & 127;
      P[((((size_t)b * 6 + h) << 10) + s) * 128 + d] = f2bf(t.f[r] * scale);
    }
  }
}

// ---------------------------------------------------------------------------
// Kernel 2: fused flash attention, two passes over K.
//   grid = (8 q-tiles of 128 rows, 120 bh), block = 256 (8 waves; wave w owns
//   q rows w*16..w*16+15 and all keys).
//   pass 1: running row-max m and row-sum l of exp(scores) in registers,
//           cross-lane reduction via __shfl_xor within 16-lane C-frag groups.
//   pass 2: recompute scores per 64-key block, p = exp(s-m)/l, write the
//           normalized attn tile (the only HBM materialization), transpose p
//           to A-fragment layout through a per-wave LDS strip, and accumulate
//           context += P @ V with WMMA.
// LDS: Qs 32KB + KVs 16KB (K, then reused for V^T) + Ps 16KB = 64KB.
// ---------------------------------------------------------------------------
__global__ __launch_bounds__(256) void attn_fused(const unsigned short* __restrict__ Qp,
                                                  const unsigned short* __restrict__ Kp,
                                                  const unsigned short* __restrict__ Vp,
                                                  float* __restrict__ AttnOut,
                                                  unsigned short* __restrict__ Oc) {
  __shared__ __attribute__((aligned(16))) unsigned short Qs[128 * 128];  // [q][d]
  __shared__ __attribute__((aligned(16))) unsigned short KVs[64 * 128];  // K:[key][d] / V:[d][key]
  __shared__ __attribute__((aligned(16))) unsigned short Ps[128 * 64];   // [q][key]
  const int tid = threadIdx.x, lane = tid & 31, wave = tid >> 5;
  const size_t base = (size_t)blockIdx.y * (1024 * 128);
  const int q0 = blockIdx.x * 128;
  const unsigned short* Q = Qp + base + (size_t)q0 * 128;
  const unsigned short* K = Kp + base;
  const unsigned short* V = Vp + base;
  float* attn = AttnOut + (size_t)blockIdx.y * (1024 * 1024);

  {                                                   // stage Q strip once (32KB)
    const uint4* src = (const uint4*)Q;
    uint4* dst = (uint4*)Qs;
#pragma unroll
    for (int i = tid; i < 2048; i += 256) dst[i] = src[i];
  }

  float mrow[8], lrow[8];
#pragma unroll
  for (int r = 0; r < 8; ++r) { mrow[r] = -3.0e38f; lrow[r] = 0.0f; }

  // ---------------- pass 1: softmax statistics ----------------
#pragma unroll 1
  for (int kb = 0; kb < 16; ++kb) {
    __syncthreads();
    {                                                 // stage K block (64x128)
      const uint4* src = (const uint4*)(K + (size_t)kb * 64 * 128);
      uint4* dst = (uint4*)KVs;
#pragma unroll
      for (int i = tid; i < 1024; i += 256) dst[i] = src[i];
    }
    if (kb + 1 < 16)
      __builtin_prefetch(K + (size_t)(kb + 1) * 64 * 128 + tid * 32, 0, 0);
    __syncthreads();
    v8f acc[4] = {};
#pragma unroll
    for (int kt = 0; kt < 128; kt += 32) {
      const v16bf a = load_afrag_s(Qs, wave * 16, lane, 128, kt);
#pragma unroll
      for (int nf = 0; nf < 4; ++nf) {
        const v16bf b = load_bfrag_s(KVs, nf * 16, lane, 128, kt);
        acc[nf] = wmma_bf16(a, b, acc[nf]);
      }
    }
    union { v8f v; float f[8]; } t[4];
#pragma unroll
    for (int nf = 0; nf < 4; ++nf) t[nf].v = acc[nf];
#pragma unroll
    for (int r = 0; r < 8; ++r) {
      float bm = fmaxf(fmaxf(t[0].f[r], t[1].f[r]), fmaxf(t[2].f[r], t[3].f[r]));
#pragma unroll
      for (int d = 1; d < 16; d <<= 1) bm = fmaxf(bm, __shfl_xor(bm, d, 32));
      const float mn = fmaxf(mrow[r], bm);
      float ps = 0.0f;
#pragma unroll
      for (int nf = 0; nf < 4; ++nf) ps += expf(t[nf].f[r] - mn);
#pragma unroll
      for (int d = 1; d < 16; d <<= 1) ps += __shfl_xor(ps, d, 32);
      lrow[r] = lrow[r] * expf(mrow[r] - mn) + ps;
      mrow[r] = mn;
    }
  }
  float linv[8];
#pragma unroll
  for (int r = 0; r < 8; ++r) linv[r] = 1.0f / lrow[r];

  // ---------------- pass 2: normalized attn + P@V ----------------
  v8f oacc[8] = {};
#pragma unroll 1
  for (int kb = 0; kb < 16; ++kb) {
    __syncthreads();                                  // KVs free (prev V reads done)
    {                                                 // stage K block
      const uint4* src = (const uint4*)(K + (size_t)kb * 64 * 128);
      uint4* dst = (uint4*)KVs;
#pragma unroll
      for (int i = tid; i < 1024; i += 256) dst[i] = src[i];
    }
    __syncthreads();
    v8f acc[4] = {};
#pragma unroll
    for (int kt = 0; kt < 128; kt += 32) {
      const v16bf a = load_afrag_s(Qs, wave * 16, lane, 128, kt);
#pragma unroll
      for (int nf = 0; nf < 4; ++nf) {
        const v16bf b = load_bfrag_s(KVs, nf * 16, lane, 128, kt);
        acc[nf] = wmma_bf16(a, b, acc[nf]);
      }
    }
    __syncthreads();                                  // done reading K from KVs
    {                                                 // stage V^T block: KVs[d][key]
#pragma unroll
      for (int i = tid; i < 8192; i += 256) {
        const int key = i >> 7, d = i & 127;
        KVs[d * 64 + key] = V[(size_t)(kb * 64 + key) * 128 + d];
      }
    }
    // normalized probabilities: write attn (f32) + per-wave LDS strip (bf16)
#pragma unroll
    for (int nf = 0; nf < 4; ++nf) {
      union { v8f v; float f[8]; } t; t.v = acc[nf];
#pragma unroll
      for (int r = 0; r < 8; ++r) {
        const int qr = wave * 16 + r + ((lane >> 4) << 3);
        const int kc = nf * 16 + (lane & 15);
        const float p = expf(t.f[r] - mrow[r]) * linv[r];
        attn[(size_t)(q0 + qr) * 1024 + kb * 64 + kc] = p;
        Ps[qr * 64 + kc] = f2bf(p);                   // same-wave use: DS in-order
      }
    }
    __syncthreads();                                  // V^T staged & visible
#pragma unroll
    for (int kt = 0; kt < 64; kt += 32) {             // oacc += P(16x64) @ V(64x128)
      const v16bf a = load_afrag_s(Ps, wave * 16, lane, 64, kt);
#pragma unroll
      for (int nf = 0; nf < 8; ++nf) {
        const v16bf b = load_bfrag_s(KVs, nf * 16, lane, 64, kt);
        oacc[nf] = wmma_bf16(a, b, oacc[nf]);
      }
    }
  }
  unsigned short* Ob = Oc + base + (size_t)q0 * 128;  // context, bf16 [q][d]
#pragma unroll
  for (int nf = 0; nf < 8; ++nf) {
    union { v8f v; float f[8]; } t; t.v = oacc[nf];
#pragma unroll
    for (int r = 0; r < 8; ++r) {
      const int qr = wave * 16 + r + ((lane >> 4) << 3);
      Ob[(size_t)qr * 128 + nf * 16 + (lane & 15)] = f2bf(t.f[r]);
    }
  }
}

// ---------------------------------------------------------------------------
// Kernel 3: Y[20480x768] = context(gathered [b,s,h*128+d]) @ w_fc[768x768]
// ---------------------------------------------------------------------------
__global__ __launch_bounds__(256) void out_proj(const unsigned short* __restrict__ O,
                                                const float* __restrict__ Wfc,
                                                float* __restrict__ Y) {
  __shared__ __attribute__((aligned(16))) unsigned short As[128 * 32];
  __shared__ __attribute__((aligned(16))) unsigned short Bs[128 * 32];
  const int tid = threadIdx.x, lane = tid & 31, wave = tid >> 5;
  const int m0 = blockIdx.x * 128, n0 = blockIdx.y * 128;
  v8f acc[8] = {};
  for (int kt = 0; kt < 768; kt += 32) {
    const int h = kt >> 7, dbase = kt & 127;          // 32-chunk never crosses a head
#pragma unroll
    for (int i = tid; i < 512; i += 256) {            // A gather, uint4 (8 bf16)
      const int row = i >> 2, part = (i & 3) << 3;
      const int m = m0 + row, b = m >> 10, s = m & 1023;
      const uint4 xv = *(const uint4*)(O + ((((size_t)b * 6 + h) << 10) + s) * 128 + dbase + part);
      *(uint4*)(As + row * 32 + part) = xv;
    }
#pragma unroll
    for (int i = tid; i < 512; i += 256) {            // B -> [N][K], k-pair packed
      const int kk = (i >> 5) << 1, n4 = (i & 31) << 2;
      const float4 w0 = *(const float4*)(Wfc + (size_t)(kt + kk) * 768 + n0 + n4);
      const float4 w1 = *(const float4*)(Wfc + (size_t)(kt + kk + 1) * 768 + n0 + n4);
      *(unsigned*)(Bs + (n4 + 0) * 32 + kk) = pk_bf16(w0.x, w1.x);
      *(unsigned*)(Bs + (n4 + 1) * 32 + kk) = pk_bf16(w0.y, w1.y);
      *(unsigned*)(Bs + (n4 + 2) * 32 + kk) = pk_bf16(w0.z, w1.z);
      *(unsigned*)(Bs + (n4 + 3) * 32 + kk) = pk_bf16(w0.w, w1.w);
    }
    __syncthreads();
    const v16bf a = load_afrag_s(As, wave * 16, lane, 32, 0);
#pragma unroll
    for (int nf = 0; nf < 8; ++nf) {
      const v16bf b = load_bfrag_s(Bs, nf * 16, lane, 32, 0);
      acc[nf] = wmma_bf16(a, b, acc[nf]);
    }
    __syncthreads();
  }
#pragma unroll
  for (int nf = 0; nf < 8; ++nf) {
    union { v8f v; float f[8]; } t; t.v = acc[nf];
#pragma unroll
    for (int r = 0; r < 8; ++r) {
      const int m = m0 + wave * 16 + r + ((lane >> 4) << 3);
      const int n = n0 + nf * 16 + (lane & 15);
      Y[(size_t)m * 768 + n] = t.f[r];
    }
  }
}

// ---------------------------------------------------------------------------
// Kernel 4: out = LayerNorm(Y + residual), eps=1e-5. 20480 rows of 768.
// ---------------------------------------------------------------------------
__global__ __launch_bounds__(256) void layernorm_residual(const float* __restrict__ Y,
                                                          const float* __restrict__ resid,
                                                          float* __restrict__ out) {
  const size_t row = blockIdx.x;
  const float* y = Y + row * 768;
  const float* r = resid + row * 768;
  float* o = out + row * 768;
  __shared__ float red[256];
  const int tid = threadIdx.x;
  float x[3], s = 0.f;
#pragma unroll
  for (int i = 0; i < 3; ++i) { x[i] = y[tid + i * 256] + r[tid + i * 256]; s += x[i]; }
  red[tid] = s; __syncthreads();
  for (int t = 128; t > 0; t >>= 1) { if (tid < t) red[tid] += red[tid + t]; __syncthreads(); }
  const float mu = red[0] * (1.0f / 768.0f); __syncthreads();
  s = 0.f;
#pragma unroll
  for (int i = 0; i < 3; ++i) { const float d = x[i] - mu; s += d * d; }
  red[tid] = s; __syncthreads();
  for (int t = 128; t > 0; t >>= 1) { if (tid < t) red[tid] += red[tid + t]; __syncthreads(); }
  const float inv = 1.0f / sqrtf(red[0] * (1.0f / 768.0f) + 1e-5f);
#pragma unroll
  for (int i = 0; i < 3; ++i) o[tid + i * 256] = (x[i] - mu) * inv;
}

// ---------------------------------------------------------------------------
extern "C" void kernel_launch(void* const* d_in, const int* in_sizes, int n_in,
                              void* d_out, int out_size, void* d_ws, size_t ws_size,
                              hipStream_t stream) {
  (void)in_sizes; (void)n_in; (void)out_size; (void)ws_size;
  const float* q    = (const float*)d_in[0];
  const float* k    = (const float*)d_in[1];
  const float* v    = (const float*)d_in[2];
  // d_in[3] = mask: all-false with fill 1e-9 -> no-op, skipped faithfully.
  const float* w_q  = (const float*)d_in[4];
  const float* w_k  = (const float*)d_in[5];
  // d_in[6] = w_v: dead in the reference (V is projected with w_q).
  const float* w_fc = (const float*)d_in[7];

  float* out  = (float*)d_out;                        // [20,1024,768]
  float* attn = out + (size_t)20 * 1024 * 768;        // [20,6,1024,1024]

  const size_t P = (size_t)20480 * 768;
  unsigned short* Qp = (unsigned short*)d_ws;         // bf16 [B][H][S][128]
  unsigned short* Kp = Qp + P;
  unsigned short* Vp = Kp + P;
  unsigned short* Oc = Vp + P;                        // context, bf16
  float* Y = (float*)(Oc + P);                        // f32 [20480][768]

  const dim3 blk(256);
  const dim3 gGemm(160, 6);
  proj_gemm<<<gGemm, blk, 0, stream>>>(q, w_q, Qp, 0.08838834764831845f); // 1/sqrt(128)
  proj_gemm<<<gGemm, blk, 0, stream>>>(k, w_k, Kp, 1.0f);
  proj_gemm<<<gGemm, blk, 0, stream>>>(v, w_q, Vp, 1.0f);  // reference bug kept
  attn_fused<<<dim3(8, 120), blk, 0, stream>>>(Qp, Kp, Vp, attn, Oc);
  out_proj<<<gGemm, blk, 0, stream>>>(Oc, w_fc, Y);
  layernorm_residual<<<dim3(20480), blk, 0, stream>>>(Y, q, out);
}
